// GCN_SAt_layer_53403623358621
// MI455X (gfx1250) — matrix operand order repaired and checked by reading, not compiled, not run on hardware
//
#include <hip/hip_runtime.h>

typedef __attribute__((ext_vector_type(16))) __bf16 v16bf;
typedef __attribute__((ext_vector_type(8)))  __bf16 v8bf;
typedef __attribute__((ext_vector_type(8)))  float  v8f;
typedef int v4i_ __attribute__((__vector_size__(16)));

static constexpr int N = 16384;   // nodes
static constexpr int F = 256;     // F_IN == F_OUT

// ---------------------------------------------------------------------------
// CDNA5 async global->LDS copy (ASYNCcnt-tracked), 16B per lane per issue.
// ---------------------------------------------------------------------------
__device__ __forceinline__ void async_copy_b128(const void* g, void* lds) {
#if __has_builtin(__builtin_amdgcn_global_load_async_to_lds_b128)
  __builtin_amdgcn_global_load_async_to_lds_b128(
      (__attribute__((address_space(1))) v4i_*)(uintptr_t)g,
      (__attribute__((address_space(3))) v4i_*)(uint32_t)(uintptr_t)lds,
      0, 0);
#else
  asm volatile("global_load_async_to_lds_b128 %0, %1, off"
               :: "v"((uint32_t)(uintptr_t)lds), "v"((uint64_t)(uintptr_t)g)
               : "memory");
#endif
}

__device__ __forceinline__ void wait_async0() {
#if __has_builtin(__builtin_amdgcn_s_wait_asynccnt)
  __builtin_amdgcn_s_wait_asynccnt(0);
#else
  asm volatile("s_wait_asynccnt 0" ::: "memory");
#endif
}

// ---------------------------------------------------------------------------
// K1: d[i] = sum_j A[i,j] + 1 ;  dinv[i] = rsqrt(d[i])
// ---------------------------------------------------------------------------
__global__ __launch_bounds__(256) void k_rowsum(const float* __restrict__ A,
                                                float* __restrict__ dinv) {
  const int row = blockIdx.x;
  const float4* Ar = reinterpret_cast<const float4*>(A + (size_t)row * N);
  float s = 0.f;
  #pragma unroll 4
  for (int c = threadIdx.x; c < N / 4; c += 256) {
    float4 v = Ar[c];
    s += (v.x + v.y) + (v.z + v.w);
  }
  __shared__ float red[256];
  red[threadIdx.x] = s;
  __syncthreads();
  #pragma unroll
  for (int off = 128; off > 0; off >>= 1) {
    if (threadIdx.x < off) red[threadIdx.x] += red[threadIdx.x + off];
    __syncthreads();
  }
  if (threadIdx.x == 0) dinv[row] = rsqrtf(red[0] + 1.0f);
}

// ---------------------------------------------------------------------------
// K2: SFT[f][j] = bf16( dinv[j] * feat[j][f] )   (transposed, bf16)
// ---------------------------------------------------------------------------
__global__ __launch_bounds__(256) void k_scale_T(const float* __restrict__ feat,
                                                 const float* __restrict__ dinv,
                                                 __bf16* __restrict__ SFT) {
  __shared__ __bf16 t[32][33];
  const int j0 = blockIdx.x * 32, f0 = blockIdx.y * 32;
  const int c = threadIdx.x & 31, r = threadIdx.x >> 5;
  #pragma unroll
  for (int i = 0; i < 4; i++) {
    const int j = j0 + r + 8 * i;
    t[r + 8 * i][c] = (__bf16)(dinv[j] * feat[(size_t)j * F + f0 + c]);
  }
  __syncthreads();
  #pragma unroll
  for (int i = 0; i < 4; i++) {
    const int f = f0 + r + 8 * i;
    SFT[(size_t)f * N + j0 + c] = t[c][r + 8 * i];
  }
}

// ---------------------------------------------------------------------------
// K2b: WT[n][k] = bf16( W[k][n] )
// ---------------------------------------------------------------------------
__global__ __launch_bounds__(256) void k_WT(const float* __restrict__ W,
                                            __bf16* __restrict__ WT) {
  __shared__ __bf16 t[32][33];
  const int k0 = blockIdx.x * 32, n0 = blockIdx.y * 32;
  const int c = threadIdx.x & 31, r = threadIdx.x >> 5;
  #pragma unroll
  for (int i = 0; i < 4; i++)
    t[r + 8 * i][c] = (__bf16)W[(size_t)(k0 + r + 8 * i) * F + n0 + c];
  __syncthreads();
  #pragma unroll
  for (int i = 0; i < 4; i++)
    WT[(size_t)(n0 + r + 8 * i) * F + k0 + c] = t[c][r + 8 * i];
}

// ---------------------------------------------------------------------------
// K3: FA[i][f] = bf16( dinv[i]*(A @ SFT^T)[i][f] + dinv[i]^2 * feat[i][f] )
// Block = 8 waves as 4(M) x 2(N); wave tile = 16(M) x 128(N), 8 accumulators.
// A: HBM stream read once, converted f32->bf16 in VALU (co-executes with XDL).
// B: 16KB chunk per block, async-DMA'd into double-buffered LDS; fragments
//    via ds_load_b128 with an interleaved two-buffer register pipeline.
// ---------------------------------------------------------------------------
#define BSTRIDE 40   // LDS row stride in bf16 (80B = 20 banks, spreads 32B reads)
__global__ __launch_bounds__(256) void k_gemm1(const float* __restrict__ A,
                                               const __bf16* __restrict__ SFT,
                                               const float* __restrict__ dinv,
                                               const float* __restrict__ feat,
                                               __bf16* __restrict__ FA) {
  __shared__ __bf16 Bt[2][256][BSTRIDE];   // 2 x 20KB

  const int tid  = threadIdx.x;
  const int wave = tid >> 5;
  const int wrow = wave >> 1;      // 0..3 : M position within block
  const int wcol = wave & 1;       // 0..1 : N position within block
  const int lane = tid & 31;
  const int lhi  = lane >> 4;      // which K half this lane supplies
  const int lm   = lane & 15;
  const int m0   = blockIdx.x * 64 + wrow * 16;
  const int nb   = wcol * 128;     // wave's N base (8 tiles of 16)
  const size_t arow = (size_t)(m0 + lm) * N;

  v8f acc[8] = {};

  // 256 threads stage 256 rows x 32 k (16KB): 1 row/thread, 4 x b128 per row.
  const int srow = tid;

  // Prologue: stage chunk 0 into buffer 0.
  #pragma unroll
  for (int q = 0; q < 4; q++)
    async_copy_b128(SFT + (size_t)srow * N + q * 8, &Bt[0][srow][q * 8]);

  for (int k0 = 0; k0 < N; k0 += 32) {
    const int buf = (k0 >> 5) & 1;

    wait_async0();         // my async writes for `buf` have landed in LDS
    __syncthreads();       // all writes landed; all waves done reading buf^1
                           // (this same barrier next iter protects re-staging)

    if (k0 + 32 < N) {     // stage next chunk into the other buffer (overlaps WMMAs)
      const int kn = k0 + 32;
      #pragma unroll
      for (int q = 0; q < 4; q++)
        async_copy_b128(SFT + (size_t)srow * N + kn + q * 8,
                        &Bt[buf ^ 1][srow][q * 8]);
    }

    // A fragment: f32 stream -> bf16 (packed cvt), in contiguous k = 16h + s order, same as B.
    const float* ap = A + arow + k0 + lhi * 16;
    __builtin_prefetch(ap + 256, 0, 1);
    v8f alo = *reinterpret_cast<const v8f*>(ap);        // K k0+16h..+7  (this half)
    v8f ahi = *reinterpret_cast<const v8f*>(ap + 8);    // K k0+16h+8..+15
    v16bf af;
    #pragma unroll
    for (int i = 0; i < 8; i++) {
      af[i]     = (__bf16)alo[i];
      af[i + 8] = (__bf16)ahi[i];
    }

    // B fragments: interleaved two-buffer pipeline over 8 n-tiles
    // (ba serves even tiles, bb odd tiles -> two live buffers, partial waits).
    const __bf16* bbase = &Bt[buf][nb + lm][lhi * 16];  // + t*16*BSTRIDE per tile
    v16bf ba = *reinterpret_cast<const v16bf*>(bbase);
    v16bf bb = *reinterpret_cast<const v16bf*>(bbase + 16 * BSTRIDE);
    #pragma unroll
    for (int t = 0; t < 8; t += 2) {
      acc[t] = __builtin_amdgcn_wmma_f32_16x16x32_bf16(
          false, af, false, ba, (short)0, acc[t], false, false);
      if (t + 2 < 8)
        ba = *reinterpret_cast<const v16bf*>(bbase + (t + 2) * 16 * BSTRIDE);
      acc[t + 1] = __builtin_amdgcn_wmma_f32_16x16x32_bf16(
          false, af, false, bb, (short)0, acc[t + 1], false, false);
      if (t + 3 < 8)
        bb = *reinterpret_cast<const v16bf*>(bbase + (t + 3) * 16 * BSTRIDE);
    }
  }

  // Epilogue: C/D layout -> VGPR r, lanes 0-15: (M=r, N=lane); lanes 16-31: (M=r+8)
  #pragma unroll
  for (int r = 0; r < 8; r++) {
    const int row  = m0 + r + lhi * 8;
    const float di = dinv[row];
    const float d2 = di * di;
    #pragma unroll
    for (int t = 0; t < 8; t++) {
      const int col = nb + t * 16 + lm;
      const float v = di * acc[t][r] + d2 * feat[(size_t)row * F + col];
      FA[(size_t)row * F + col] = (__bf16)v;
    }
  }
}

// ---------------------------------------------------------------------------
// K4: out = l2normalize_rows( relu( FA @ W ) )
// ---------------------------------------------------------------------------
__global__ __launch_bounds__(256) void k_gemm2_norm(const __bf16* __restrict__ FA,
                                                    const __bf16* __restrict__ WT,
                                                    float* __restrict__ out) {
  const int wave = threadIdx.x >> 5;
  const int lane = threadIdx.x & 31;
  const int lhi = lane >> 4, lm = lane & 15;
  const int m0 = blockIdx.x * 16;

  v8f acc[2] = {};
  #pragma unroll
  for (int k0 = 0; k0 < F; k0 += 32) {
    const __bf16* ap = FA + (size_t)(m0 + lm) * F + k0 + lhi * 16;
    v8bf lo = *reinterpret_cast<const v8bf*>(ap);
    v8bf hi = *reinterpret_cast<const v8bf*>(ap + 8);
    v16bf af;
    #pragma unroll
    for (int i = 0; i < 8; i++) { af[i] = lo[i]; af[i + 8] = hi[i]; }
    #pragma unroll
    for (int t = 0; t < 2; t++) {
      const int n = wave * 32 + t * 16 + lm;
      v16bf bf_ = *reinterpret_cast<const v16bf*>(WT + (size_t)n * F + k0 + lhi * 16);
      acc[t] = __builtin_amdgcn_wmma_f32_16x16x32_bf16(
          false, af, false, bf_, (short)0, acc[t], false, false);
    }
  }

  __shared__ float sm[16][257];
  #pragma unroll
  for (int t = 0; t < 2; t++)
    #pragma unroll
    for (int r = 0; r < 8; r++)
      sm[r + lhi * 8][wave * 32 + t * 16 + lm] = fmaxf(acc[t][r], 0.f);
  __syncthreads();

  const int rr = threadIdx.x >> 4, cc = threadIdx.x & 15;
  float s = 0.f;
  #pragma unroll
  for (int i = 0; i < 16; i++) { const float v = sm[rr][cc + 16 * i]; s += v * v; }
  __shared__ float ps[16][17];
  ps[rr][cc] = s;
  __syncthreads();
  __shared__ float scl[16];
  if (threadIdx.x < 16) {
    float ss = 0.f;
    #pragma unroll
    for (int i = 0; i < 16; i++) ss += ps[threadIdx.x][i];
    scl[threadIdx.x] = 1.f / fmaxf(sqrtf(ss), 1e-12f);
  }
  __syncthreads();
  #pragma unroll
  for (int i = 0; i < 16; i++)
    out[(size_t)(m0 + i) * F + threadIdx.x] = sm[i][threadIdx.x] * scl[i];
}

// ---------------------------------------------------------------------------
// Workspace layout (bytes): dinv[64K] | SFT[8M] | WT[128K] | FA[8M]  (~16.6 MB)
// ---------------------------------------------------------------------------
extern "C" void kernel_launch(void* const* d_in, const int* in_sizes, int n_in,
                              void* d_out, int out_size, void* d_ws, size_t ws_size,
                              hipStream_t stream) {
  const float* feat = (const float*)d_in[0];
  const float* A    = (const float*)d_in[1];
  const float* W    = (const float*)d_in[2];
  float* out = (float*)d_out;

  char* ws = (char*)d_ws;
  float*  dinv = (float*)ws;                                   // 16384 f32
  __bf16* SFT  = (__bf16*)(ws + 64 * 1024);                    // 256 x 16384 bf16
  __bf16* WT   = (__bf16*)(ws + 64 * 1024 + 8 * 1024 * 1024);  // 256 x 256 bf16
  __bf16* FA   = (__bf16*)(ws + 64 * 1024 + 8 * 1024 * 1024 + 128 * 1024); // 16384 x 256

  k_rowsum   <<<N, 256, 0, stream>>>(A, dinv);
  k_scale_T  <<<dim3(N / 32, F / 32), 256, 0, stream>>>(feat, dinv, SFT);
  k_WT       <<<dim3(F / 32, F / 32), 256, 0, stream>>>(W, WT);
  k_gemm1    <<<N / 64, 256, 0, stream>>>(A, SFT, dinv, feat, FA);
  k_gemm2_norm<<<N / 16, 256, 0, stream>>>(FA, WT, out);
}
